// MarginLoss_52682068853382
// MI455X (gfx1250) — compile-verified, matrix operand order checked by reading, not run on hardware
//
#include <hip/hip_runtime.h>
#include <hip/hip_bf16.h>
#include <math.h>

#define BS       4096
#define DIM      256
#define NT32     128          // 4096 / 32 (fallback tiling)
#define NT64     64           // 4096 / 64 (main tiling)
#define MARGIN   0.2f

typedef __attribute__((ext_vector_type(16))) __bf16 v16bf;
typedef __attribute__((ext_vector_type(8)))  float  v8f;

struct Frag { v16bf hi; v16bf lo; };

// ---- bf16x3 fp32-accurate accumulate: C += Ahi*Bhi + Ahi*Blo + Alo*Bhi ----
__device__ __forceinline__ v8f wmma3(const Frag& a, const Frag& b, v8f c) {
    c = __builtin_amdgcn_wmma_f32_16x16x32_bf16(false, a.hi, false, b.hi, (short)0, c, false, false);
    c = __builtin_amdgcn_wmma_f32_16x16x32_bf16(false, a.hi, false, b.lo, (short)0, c, false, false);
    c = __builtin_amdgcn_wmma_f32_16x16x32_bf16(false, a.lo, false, b.hi, (short)0, c, false, false);
    return c;
}

// Fragment layout (ISA 7.12.2, 16-bit A 16x32): lane m = L&15 -> row; half = L>>4:
//   vector elems 0..7  = K + 8*half + (0..7)
//   vector elems 8..15 = K + 16 + 8*half + (0..7)
union FragHalf { v16bf v; uint4 q[2]; };

__device__ __forceinline__ v16bf load_plane(const __bf16* __restrict__ row, int kbase, int half) {
    FragHalf u;
    u.q[0] = *reinterpret_cast<const uint4*>(row + kbase + 8 * half);
    u.q[1] = *reinterpret_cast<const uint4*>(row + kbase + 16 + 8 * half);
    return u.v;
}

__device__ __forceinline__ Frag load_frag_pre(const __bf16* __restrict__ hi,
                                              const __bf16* __restrict__ lo,
                                              size_t rowoff, int kbase, int half) {
    Frag r;
    r.hi = load_plane(hi + rowoff, kbase, half);
    r.lo = load_plane(lo + rowoff, kbase, half);
    return r;
}

// On-the-fly fp32 -> hi/lo fragment (fallback path only)
__device__ __forceinline__ Frag load_frag_fly(const float* __restrict__ Xrow, int kbase, int half) {
    float f[16];
    const float4* p0 = reinterpret_cast<const float4*>(Xrow + kbase + 8 * half);
    const float4* p1 = reinterpret_cast<const float4*>(Xrow + kbase + 16 + 8 * half);
    float4 t;
    t = p0[0]; f[0] = t.x; f[1] = t.y; f[2]  = t.z; f[3]  = t.w;
    t = p0[1]; f[4] = t.x; f[5] = t.y; f[6]  = t.z; f[7]  = t.w;
    t = p1[0]; f[8] = t.x; f[9] = t.y; f[10] = t.z; f[11] = t.w;
    t = p1[1]; f[12] = t.x; f[13] = t.y; f[14] = t.z; f[15] = t.w;
    Frag r;
#pragma unroll
    for (int e = 0; e < 16; ++e) {
        __bf16 h = (__bf16)f[e];
        r.hi[e] = h;
        r.lo[e] = (__bf16)(f[e] - (float)h);
    }
    return r;
}

// ---------------- kernel 0: init workspace ----------------
__global__ void ml_init(unsigned int* __restrict__ dmin, float* __restrict__ acc) {
    int i = blockIdx.x * blockDim.x + threadIdx.x;
    if (i < BS) dmin[i] = 0x7F800000u;           // +inf
    if (i == 0) { acc[0] = 0.f; acc[1] = 0.f; acc[2] = 0.f; }
}

// ---------------- kernel 1: exact fp32 row squared norms ----------------
__global__ void ml_sqnorm(const float* __restrict__ X, float* __restrict__ sq) {
    int wid  = blockIdx.x * (blockDim.x >> 5) + (threadIdx.x >> 5);
    int lane = threadIdx.x & 31;
    if (wid >= BS) return;
    const float* row = X + (size_t)wid * DIM;
    float s = 0.f;
#pragma unroll
    for (int k = lane; k < DIM; k += 32) { float v = row[k]; s += v * v; }
#pragma unroll
    for (int off = 16; off >= 1; off >>= 1) s += __shfl_xor(s, off);
    if (lane == 0) sq[wid] = s;
}

// ---------------- kernel 1b: one-pass fp32 -> bf16 hi/lo planes ----------------
__global__ void ml_split(const float* __restrict__ X, __bf16* __restrict__ xhi,
                         __bf16* __restrict__ xlo) {
    int i = blockIdx.x * blockDim.x + threadIdx.x;
    if (i >= BS * DIM) return;
    float v = X[i];
    __bf16 h = (__bf16)v;
    xhi[i] = h;
    xlo[i] = (__bf16)(v - (float)h);
}

// ---------------- kernel 2: WMMA Gram (64x64 tile/wave) + fused min ----------------
__global__ __launch_bounds__(256) void ml_gram_min64(const __bf16* __restrict__ xhi,
                                                     const __bf16* __restrict__ xlo,
                                                     const float* __restrict__ sq,
                                                     unsigned int* __restrict__ dmin) {
    const int wid  = blockIdx.x * 8 + (threadIdx.x >> 5);
    const int lane = threadIdx.x & 31;
    const int ti = wid >> 6;            // /NT64
    const int tj = wid & (NT64 - 1);
    if (tj < ti) return;                // symmetry: upper triangle only
    const int I = ti * 64, J = tj * 64;
    const int m = lane & 15, half = lane >> 4;

    v8f c[4][4] = {};
#pragma unroll 1
    for (int K = 0; K < DIM; K += 32) {
        Frag A[4], B[4];
#pragma unroll
        for (int f = 0; f < 4; ++f)
            A[f] = load_frag_pre(xhi, xlo, (size_t)(I + 16 * f + m) * DIM, K, half);
#pragma unroll
        for (int g = 0; g < 4; ++g)
            B[g] = load_frag_pre(xhi, xlo, (size_t)(J + 16 * g + m) * DIM, K, half);
#pragma unroll
        for (int f = 0; f < 4; ++f)
#pragma unroll
            for (int g = 0; g < 4; ++g)
                c[f][g] = wmma3(A[f], B[g], c[f][g]);
    }

    const float INFF = __uint_as_float(0x7F800000u);
    const int n  = lane & 15;           // C/D layout: N = lane&15
    const int mb = (lane >> 4) * 8;     // M = mb + vgpr_index
    float sqc[4], cmin[4];
#pragma unroll
    for (int g = 0; g < 4; ++g) { sqc[g] = sq[J + 16 * g + n]; cmin[g] = INFF; }

#pragma unroll
    for (int f = 0; f < 4; ++f) {
#pragma unroll
        for (int r = 0; r < 8; ++r) {
            const int row = I + 16 * f + mb + r;
            const float sqr = sq[row];
            float rm = INFF;
#pragma unroll
            for (int g = 0; g < 4; ++g) {
                float d = fmaxf(sqr + sqc[g] - 2.f * c[f][g][r], 0.f);
                if (row == J + 16 * g + n) d = INFF;   // mask diagonal
                cmin[g] = fminf(cmin[g], d);
                rm = fminf(rm, d);
            }
#pragma unroll
            for (int off = 8; off >= 1; off >>= 1) rm = fminf(rm, __shfl_xor(rm, off));
            if (n == 0) atomicMin(dmin + row, __float_as_uint(rm));  // lanes 0 & 16
        }
    }
#pragma unroll
    for (int g = 0; g < 4; ++g) {
        cmin[g] = fminf(cmin[g], __shfl_xor(cmin[g], 16));
        if (lane < 16) atomicMin(dmin + J + 16 * g + n, __float_as_uint(cmin[g]));
    }
}

// ---------------- kernel 2-fallback: 32x32 tile/wave, on-the-fly split ----------------
__global__ __launch_bounds__(256) void ml_gram_min_fly(const float* __restrict__ X,
                                                       const float* __restrict__ sq,
                                                       unsigned int* __restrict__ dmin) {
    const int wid  = blockIdx.x * 8 + (threadIdx.x >> 5);
    const int lane = threadIdx.x & 31;
    const int ti = wid >> 7;
    const int tj = wid & (NT32 - 1);
    if (tj < ti) return;
    const int I = ti * 32, J = tj * 32;
    const int m = lane & 15, half = lane >> 4;

    v8f c00 = {}, c01 = {}, c10 = {}, c11 = {};
    const float* rA0 = X + (size_t)(I + m) * DIM;
    const float* rA1 = X + (size_t)(I + 16 + m) * DIM;
    const float* rB0 = X + (size_t)(J + m) * DIM;
    const float* rB1 = X + (size_t)(J + 16 + m) * DIM;
#pragma unroll
    for (int K = 0; K < DIM; K += 32) {
        Frag a0 = load_frag_fly(rA0, K, half);
        Frag a1 = load_frag_fly(rA1, K, half);
        Frag b0 = load_frag_fly(rB0, K, half);
        Frag b1 = load_frag_fly(rB1, K, half);
        c00 = wmma3(a0, b0, c00);
        c01 = wmma3(a0, b1, c01);
        c10 = wmma3(a1, b0, c10);
        c11 = wmma3(a1, b1, c11);
    }
    const float INFF = __uint_as_float(0x7F800000u);
    const int n  = lane & 15;
    const int mb = (lane >> 4) * 8;
    const float sqc0 = sq[J + n], sqc1 = sq[J + 16 + n];
    float cmin0 = INFF, cmin1 = INFF;
#pragma unroll
    for (int r = 0; r < 8; ++r) {
        const int row0 = I + mb + r, row1 = I + 16 + mb + r;
        const float sqr0 = sq[row0], sqr1 = sq[row1];
        float d00 = fmaxf(sqr0 + sqc0 - 2.f * c00[r], 0.f); if (row0 == J + n)      d00 = INFF;
        float d01 = fmaxf(sqr0 + sqc1 - 2.f * c01[r], 0.f); if (row0 == J + 16 + n) d01 = INFF;
        float d10 = fmaxf(sqr1 + sqc0 - 2.f * c10[r], 0.f); if (row1 == J + n)      d10 = INFF;
        float d11 = fmaxf(sqr1 + sqc1 - 2.f * c11[r], 0.f); if (row1 == J + 16 + n) d11 = INFF;
        cmin0 = fminf(cmin0, fminf(d00, d10));
        cmin1 = fminf(cmin1, fminf(d01, d11));
        float rm0 = fminf(d00, d01), rm1 = fminf(d10, d11);
#pragma unroll
        for (int off = 8; off >= 1; off >>= 1) {
            rm0 = fminf(rm0, __shfl_xor(rm0, off));
            rm1 = fminf(rm1, __shfl_xor(rm1, off));
        }
        if (n == 0) {
            atomicMin(dmin + row0, __float_as_uint(rm0));
            atomicMin(dmin + row1, __float_as_uint(rm1));
        }
    }
    cmin0 = fminf(cmin0, __shfl_xor(cmin0, 16));
    cmin1 = fminf(cmin1, __shfl_xor(cmin1, 16));
    if (lane < 16) {
        atomicMin(dmin + J + n,      __float_as_uint(cmin0));
        atomicMin(dmin + J + 16 + n, __float_as_uint(cmin1));
    }
}

// ---------------- kernel 3: triplet losses + log-min sum ----------------
__global__ void ml_triplet(const float* __restrict__ X, const float* __restrict__ beta,
                           const int* __restrict__ labels, const int* __restrict__ pos,
                           const int* __restrict__ neg, const unsigned int* __restrict__ dmin,
                           float* __restrict__ acc) {
    int wid  = blockIdx.x * (blockDim.x >> 5) + (threadIdx.x >> 5);
    int lane = threadIdx.x & 31;
    if (wid >= BS) return;
    const float* xi = X + (size_t)wid * DIM;
    const float* xp = X + (size_t)pos[wid] * DIM;
    const float* xn = X + (size_t)neg[wid] * DIM;
    float dap = 0.f, dan = 0.f;
#pragma unroll
    for (int k = lane; k < DIM; k += 32) {
        float a = xi[k];
        float dp = a - xp[k]; dap += dp * dp;
        float dn = a - xn[k]; dan += dn * dn;
    }
#pragma unroll
    for (int off = 16; off >= 1; off >>= 1) { dap += __shfl_xor(dap, off); dan += __shfl_xor(dan, off); }
    if (lane == 0) {
        float d_ap = sqrtf(dap + 1e-8f);
        float d_an = sqrtf(dan + 1e-8f);
        float b  = beta[labels[wid]];
        float pl = fmaxf(d_ap - b + MARGIN, 0.f);
        float nl = fmaxf(b - d_an + MARGIN, 0.f);
        atomicAdd(acc + 0, pl + nl);
        atomicAdd(acc + 1, (pl > 0.f || nl > 0.f) ? 1.f : 0.f);
        atomicAdd(acc + 2, 0.5f * logf(__uint_as_float(dmin[wid])));  // log(sqrt(min d^2))
    }
}

// ---------------- kernel 4: finalize ----------------
__global__ void ml_finalize(const float* __restrict__ acc, float* __restrict__ out) {
    if (threadIdx.x == 0 && blockIdx.x == 0) {
        float total = acc[0], cnt = acc[1], logsum = acc[2];
        float loss = (cnt == 0.f) ? total : (total / fmaxf(cnt, 1.f));
        out[0] = loss - logsum;
    }
}

extern "C" void kernel_launch(void* const* d_in, const int* in_sizes, int n_in,
                              void* d_out, int out_size, void* d_ws, size_t ws_size,
                              hipStream_t stream) {
    const float* X      = (const float*)d_in[0];
    const float* beta   = (const float*)d_in[1];
    const int*   labels = (const int*)d_in[2];
    const int*   pos    = (const int*)d_in[3];
    const int*   neg    = (const int*)d_in[4];
    float* out = (float*)d_out;

    char* ws = (char*)d_ws;
    unsigned int* d_min = (unsigned int*)(ws);                 // 16 KB
    float*        d_sq  = (float*)(ws + 16384);                // 16 KB
    float*        d_acc = (float*)(ws + 32768);                // 3 floats
    __bf16*       d_xhi = (__bf16*)(ws + 65536);               // 2 MB
    __bf16*       d_xlo = (__bf16*)(ws + 65536 + (size_t)BS * DIM * 2);  // 2 MB
    const size_t need_pre = 65536 + (size_t)BS * DIM * 2 * 2;

    ml_init<<<(BS + 255) / 256, 256, 0, stream>>>(d_min, d_acc);
    ml_sqnorm<<<BS / 8, 256, 0, stream>>>(X, d_sq);
    if (ws_size >= need_pre) {
        ml_split<<<(BS * DIM) / 256, 256, 0, stream>>>(X, d_xhi, d_xlo);
        ml_gram_min64<<<(NT64 * NT64) / 8, 256, 0, stream>>>(d_xhi, d_xlo, d_sq, d_min);
    } else {
        ml_gram_min_fly<<<(NT32 * NT32) / 8, 256, 0, stream>>>(X, d_sq, d_min);
    }
    ml_triplet<<<BS / 8, 256, 0, stream>>>(X, beta, labels, pos, neg, d_min, d_acc);
    ml_finalize<<<1, 32, 0, stream>>>(d_acc, out);
}
